// TarMAC_Actor_60662118089449
// MI455X (gfx1250) — matrix operand order repaired and checked by reading, not compiled, not run on hardware
//
#include <hip/hip_runtime.h>

typedef __bf16 bf16;
typedef __attribute__((ext_vector_type(4)))  float v4f;
typedef __attribute__((ext_vector_type(8)))  bf16  v8bf;
typedef __attribute__((ext_vector_type(16))) bf16  v16bf;
typedef __attribute__((ext_vector_type(8)))  float v8f;

// ---------------------------------------------------------------------------
// Tensor Data Mover staging (gfx1250). Guarded: falls back to vectorized copy.
// ---------------------------------------------------------------------------
#if defined(__has_builtin)
#if __has_builtin(__builtin_amdgcn_tensor_load_to_lds) && \
    __has_builtin(__builtin_amdgcn_s_wait_tensorcnt)
#define HAVE_TDM 1
#endif
#endif
#ifndef HAVE_TDM
#define HAVE_TDM 0
#endif

#if HAVE_TDM
typedef __attribute__((ext_vector_type(4))) unsigned int u32x4;
typedef __attribute__((ext_vector_type(8))) int i32x8;
typedef __attribute__((ext_vector_type(4))) int i32x4;

// DMA a dense 2D bf16 tile [rows x cols] (global row stride = gstride elems)
// into LDS at ldsDst. One issuing wave; completion via TENSORcnt.
// D# layout per CDNA5 ISA.
static __device__ __forceinline__ void tdm_load_tile_bf16(
    bf16* ldsDst, const bf16* gSrc, unsigned rows, unsigned cols,
    unsigned gstride) {
  unsigned long long ga = (unsigned long long)gSrc;
  unsigned lo = (unsigned)(unsigned long long)ldsDst;  // low 32b = LDS offset
  u32x4 g0;
  g0[0] = 1u;                                   // count=1 (valid user D#)
  g0[1] = lo;                                   // lds_addr (bytes)
  g0[2] = (unsigned)(ga & 0xffffffffu);         // global_addr[31:0]
  g0[3] = (unsigned)((ga >> 32) & 0x01ffffffu)  // global_addr[56:32]
          | (2u << 30);                         // type = 2 ("image")
  i32x8 g1;
  g1[0] = (int)(1u << 16);                      // data_size=1 -> 2 bytes
  g1[1] = (int)(cols << 16);                    // tensor_dim0[15:0]
  g1[2] = (int)((cols >> 16) | (rows << 16));   // dim0 hi | tensor_dim1 lo
  g1[3] = (int)((rows >> 16) | (cols << 16));   // dim1 hi | tile_dim0
  g1[4] = (int)(rows & 0xffffu);                // tile_dim1 (tile_dim2 = 0)
  g1[5] = (int)gstride;                         // tensor_dim0_stride[31:0]
  g1[6] = 0;                                    // stride hi | dim1_stride lo
  g1[7] = 0;
  i32x4 z = {0, 0, 0, 0};
#if __clang_major__ >= 23
  i32x8 z8 = {0, 0, 0, 0, 0, 0, 0, 0};
  __builtin_amdgcn_tensor_load_to_lds(g0, g1, z, z, z8, 0);
#else
  __builtin_amdgcn_tensor_load_to_lds(g0, g1, z, z, 0);
#endif
}
#endif  // HAVE_TDM

// ---------------------------------------------------------------------------
// Core WMMA building block: C[16x16] += A[16x32] * B[32x16] over Ksz depth.
//   A:  row-major bf16, stride lda  (rows = M)       -> A fragment per ISA
//   Bt: row-major bf16, stride ldb  (rows = N cols)  -> B fragment (Bt[n][k])
// Pointers may be LDS or global; forceinline lets clang infer address space.
// ---------------------------------------------------------------------------
static __device__ __forceinline__ v8f wmma_accum(const bf16* A, int lda,
                                                 const bf16* Bt, int ldb,
                                                 int Ksz, v8f acc) {
  const int lane = threadIdx.x & 31;
  const int hl   = lane >> 4;   // which 16-lane half
  const int mr   = lane & 15;   // row (A) / col (B)
  for (int k0 = 0; k0 < Ksz; k0 += 32) {
    union { v16bf v; v8bf h[2]; } a, b;
    const bf16* pa = A + mr * lda + k0 + hl * 8;
    a.h[0] = *(const v8bf*)(pa);          // K = k0 + 8*hl + [0..7]
    a.h[1] = *(const v8bf*)(pa + 16);     // K = k0 + 16 + 8*hl + [0..7]
    const bf16* pb = Bt + mr * ldb + k0 + hl * 8;
    b.h[0] = *(const v8bf*)(pb);          // K = k0 + 8*hl + [0..7]
    b.h[1] = *(const v8bf*)(pb + 16);     // K = k0 + 16 + 8*hl + [0..7]
    acc = __builtin_amdgcn_wmma_f32_16x16x32_bf16(false, a.v, false, b.v,
                                                  (short)0, acc, false, false);
  }
  return acc;
}

// ---------------------------------------------------------------------------
// Weight prep: fp32 W[in][out] -> bf16 Wt[out][in]
// ---------------------------------------------------------------------------
__global__ void k_transpose_bf16(const float* __restrict__ W, bf16* __restrict__ Wt,
                                 int inDim, int outDim) {
  int idx = blockIdx.x * blockDim.x + threadIdx.x;
  if (idx < inDim * outDim) {
    int r = idx / outDim, c = idx - r * outDim;
    Wt[c * inDim + r] = (bf16)W[idx];
  }
}

// ---------------------------------------------------------------------------
// Kernel 1: x = ReLU(obs @ W1 + b1) @ W2 + b2   -> bf16 x[R][256]
// 128 thr (4 waves), 64 rows per block.
// ---------------------------------------------------------------------------
__global__ __launch_bounds__(128) void k_obs2hidden(
    const float* __restrict__ obs,
    const bf16* __restrict__ Wt1, const float* __restrict__ b1,
    const bf16* __restrict__ Wt2, const float* __restrict__ b2,
    bf16* __restrict__ xout) {
  __shared__ bf16 sObs[64 * 64];    // 8 KB
  __shared__ bf16 sHid[64 * 256];   // 32 KB
  const int tid = threadIdx.x, wave = tid >> 5, lane = tid & 31;
  const int hl = lane >> 4, mr = lane & 15;
  const size_t rowBase = (size_t)blockIdx.x * 64;

  const float* g = obs + rowBase * 64;
  for (int i = tid * 8; i < 64 * 64; i += 128 * 8) {
    v4f f0 = *(const v4f*)(g + i);
    v4f f1 = *(const v4f*)(g + i + 4);
    v8bf o;
    #pragma unroll
    for (int j = 0; j < 4; ++j) { o[j] = (bf16)f0[j]; o[4 + j] = (bf16)f1[j]; }
    *(v8bf*)(sObs + i) = o;
  }
  __syncthreads();

  const bf16* aRow = sObs + wave * 16 * 64;
  #pragma unroll 4
  for (int ct = 0; ct < 16; ++ct) {                     // hidden cols 0..255
    v8f acc = {};
    acc = wmma_accum(aRow, 64, Wt1 + ct * 16 * 64, 64, 64, acc);
    const float bias = b1[ct * 16 + mr];
    #pragma unroll
    for (int r = 0; r < 8; ++r) {
      float v = acc[r] + bias;
      sHid[(wave * 16 + r + 8 * hl) * 256 + ct * 16 + mr] = (bf16)(v > 0.f ? v : 0.f);
    }
  }
  __syncthreads();

  const bf16* hRow = sHid + wave * 16 * 256;
  #pragma unroll 4
  for (int ct = 0; ct < 16; ++ct) {                     // x cols 0..255
    v8f acc = {};
    acc = wmma_accum(hRow, 256, Wt2 + ct * 16 * 256, 256, 256, acc);
    const float bias = b2[ct * 16 + mr];
    #pragma unroll
    for (int r = 0; r < 8; ++r)
      xout[(rowBase + wave * 16 + r + 8 * hl) * 256 + ct * 16 + mr] =
          (bf16)(acc[r] + bias);
  }
}

// ---------------------------------------------------------------------------
// Kernel 2: out = tanh(x @ Wp1 + bp1) @ Wp2 + bp2   -> bf16 out[R][32]
// x tile staged to LDS by the Tensor Data Mover (fallback: vector copy).
// ---------------------------------------------------------------------------
__global__ __launch_bounds__(128) void k_proj(
    const bf16* __restrict__ xin,
    const bf16* __restrict__ W1t, const float* __restrict__ b1,
    const bf16* __restrict__ W2t, const float* __restrict__ b2,
    bf16* __restrict__ out) {
  __shared__ bf16 sX[64 * 256];     // 32 KB
  __shared__ bf16 sHid[64 * 256];   // 32 KB
  const int tid = threadIdx.x, wave = tid >> 5, lane = tid & 31;
  const int hl = lane >> 4, mr = lane & 15;
  const size_t rowBase = (size_t)blockIdx.x * 64;

  const bf16* g = xin + rowBase * 256;
#if HAVE_TDM
  if (wave == 0) {
    tdm_load_tile_bf16(sX, g, 64, 256, 256);
    __builtin_amdgcn_s_wait_tensorcnt(0);
  }
#else
  for (int i = tid * 8; i < 64 * 256; i += 128 * 8)
    *(v8bf*)(sX + i) = *(const v8bf*)(g + i);
#endif
  __syncthreads();

  const bf16* aRow = sX + wave * 16 * 256;
  #pragma unroll 4
  for (int ct = 0; ct < 16; ++ct) {
    v8f acc = {};
    acc = wmma_accum(aRow, 256, W1t + ct * 16 * 256, 256, 256, acc);
    const float bias = b1[ct * 16 + mr];
    #pragma unroll
    for (int r = 0; r < 8; ++r)
      sHid[(wave * 16 + r + 8 * hl) * 256 + ct * 16 + mr] = (bf16)tanhf(acc[r] + bias);
  }
  __syncthreads();

  const bf16* hRow = sHid + wave * 16 * 256;
  #pragma unroll
  for (int ct = 0; ct < 2; ++ct) {                      // 32 output cols
    v8f acc = {};
    acc = wmma_accum(hRow, 256, W2t + ct * 16 * 256, 256, 256, acc);
    const float bias = b2[ct * 16 + mr];
    #pragma unroll
    for (int r = 0; r < 8; ++r)
      out[(rowBase + wave * 16 + r + 8 * hl) * 32 + ct * 16 + mr] =
          (bf16)(acc[r] + bias);
  }
}

// ---------------------------------------------------------------------------
// Kernel 3: per batch: scores = q k^T / sqrt(32); softmax; comm = attn v
// block = (batch, 64-row group); 128 thr; LDS = K(16K)+V^T(16K)+attn(32K)
// K matrix staged via TDM; V^T needs transpose so stays a lane copy.
// ---------------------------------------------------------------------------
__global__ __launch_bounds__(128) void k_attn(
    const bf16* __restrict__ qg, const bf16* __restrict__ kg,
    const bf16* __restrict__ vg, bf16* __restrict__ commg) {
  __shared__ bf16 sK[256 * 32];
  __shared__ bf16 sVT[32 * 256];
  __shared__ bf16 sAttn[4 * 16 * 256];
  const int tid = threadIdx.x, wave = tid >> 5, lane = tid & 31;
  const int hl = lane >> 4, mr = lane & 15;
  const int b = blockIdx.x >> 2, rg = blockIdx.x & 3;
  const size_t base = (size_t)b * 256 * 32;

#if HAVE_TDM
  if (wave == 0) tdm_load_tile_bf16(sK, kg + base, 256, 32, 32);
#else
  for (int i = tid * 8; i < 256 * 32; i += 128 * 8)
    *(v8bf*)(sK + i) = *(const v8bf*)(kg + base + i);
#endif
  for (int i = tid; i < 256 * 32; i += 128) {
    int n = i >> 5, c = i & 31;
    sVT[c * 256 + n] = vg[base + i];
  }
#if HAVE_TDM
  if (wave == 0) __builtin_amdgcn_s_wait_tensorcnt(0);
#endif
  __syncthreads();

  const int rowLocal = rg * 64 + wave * 16;
  const bf16* qRow = qg + base + (size_t)rowLocal * 32;

  v8f s[16];
  const float scale = 0.17677669529663687f;  // 1/sqrt(32)
  #pragma unroll
  for (int ct = 0; ct < 16; ++ct) {          // 16 col tiles -> 256 scores/row
    v8f acc = {};
    acc = wmma_accum(qRow, 32, sK + ct * 16 * 32, 32, 32, acc);
    #pragma unroll
    for (int r = 0; r < 8; ++r) acc[r] *= scale;
    s[ct] = acc;
  }
  // softmax across 256 cols; row (r,hl) is spread across 16 lanes of one half
  float rinv[8];
  #pragma unroll
  for (int r = 0; r < 8; ++r) {
    float m = s[0][r];
    #pragma unroll
    for (int ct = 1; ct < 16; ++ct) m = fmaxf(m, s[ct][r]);
    #pragma unroll
    for (int d = 1; d < 16; d <<= 1) m = fmaxf(m, __shfl_xor(m, d, 32));
    float sum = 0.f;
    #pragma unroll
    for (int ct = 0; ct < 16; ++ct) {
      float e = __expf(s[ct][r] - m);
      s[ct][r] = e;
      sum += e;
    }
    #pragma unroll
    for (int d = 1; d < 16; d <<= 1) sum += __shfl_xor(sum, d, 32);
    rinv[r] = 1.f / sum;
  }
  bf16* aW = sAttn + wave * 16 * 256;        // re-lay attn as A-matrix rows
  #pragma unroll
  for (int ct = 0; ct < 16; ++ct)
    #pragma unroll
    for (int r = 0; r < 8; ++r)
      aW[(r + 8 * hl) * 256 + ct * 16 + mr] = (bf16)(s[ct][r] * rinv[r]);
  __syncthreads();

  #pragma unroll
  for (int ct = 0; ct < 2; ++ct) {           // comm cols 0..31
    v8f acc = {};
    acc = wmma_accum(aW, 256, sVT + ct * 16 * 256, 256, 256, acc);
    #pragma unroll
    for (int r = 0; r < 8; ++r)
      commg[((size_t)b * 256 + rowLocal + r + 8 * hl) * 32 + ct * 16 + mr] =
          (bf16)acc[r];
  }
}

// ---------------------------------------------------------------------------
// Kernel 4: logits = ReLU([x|comm] @ Wa1 + ba1) @ Wa2 + ba2; softmax -> fp32
// 64 thr (2 waves), 32 rows per block. K = 288 = 9 WMMA steps.
// ---------------------------------------------------------------------------
__global__ __launch_bounds__(64) void k_head(
    const bf16* __restrict__ xg, const bf16* __restrict__ commg,
    const bf16* __restrict__ Wa1t, const float* __restrict__ ba1,
    const bf16* __restrict__ Wa2t, const float* __restrict__ ba2,
    float* __restrict__ outp) {
  __shared__ bf16 sH[32 * 288];     // 18 KB  concat [x | comm]
  __shared__ bf16 sHid[32 * 256];   // 16 KB
  const int tid = threadIdx.x, wave = tid >> 5, lane = tid & 31;
  const int hl = lane >> 4, mr = lane & 15;
  const size_t rowBase = (size_t)blockIdx.x * 32;

  for (int i = tid * 8; i < 32 * 256; i += 64 * 8) {
    int r = i >> 8, c = i & 255;
    *(v8bf*)(sH + r * 288 + c) = *(const v8bf*)(xg + (rowBase + r) * 256 + c);
  }
  for (int i = tid * 8; i < 32 * 32; i += 64 * 8) {
    int r = i >> 5, c = i & 31;
    *(v8bf*)(sH + r * 288 + 256 + c) = *(const v8bf*)(commg + (rowBase + r) * 32 + c);
  }
  __syncthreads();

  const bf16* hRow = sH + wave * 16 * 288;
  #pragma unroll 4
  for (int ct = 0; ct < 16; ++ct) {
    v8f acc = {};
    acc = wmma_accum(hRow, 288, Wa1t + ct * 16 * 288, 288, 288, acc);
    const float bias = ba1[ct * 16 + mr];
    #pragma unroll
    for (int r = 0; r < 8; ++r) {
      float v = acc[r] + bias;
      sHid[(wave * 16 + r + 8 * hl) * 256 + ct * 16 + mr] = (bf16)(v > 0.f ? v : 0.f);
    }
  }
  __syncthreads();

  v8f acc = {};
  acc = wmma_accum(sHid + wave * 16 * 256, 256, Wa2t, 256, 256, acc);
  const float bias = ba2[mr];
  #pragma unroll
  for (int r = 0; r < 8; ++r) {                // softmax over 16 logits/row
    float e = acc[r] + bias;
    float m = e;
    #pragma unroll
    for (int d = 1; d < 16; d <<= 1) m = fmaxf(m, __shfl_xor(m, d, 32));
    float ex = __expf(e - m);
    float sum = ex;
    #pragma unroll
    for (int d = 1; d < 16; d <<= 1) sum += __shfl_xor(sum, d, 32);
    outp[(rowBase + wave * 16 + r + 8 * hl) * 16 + mr] = ex / sum;
  }
}

// ---------------------------------------------------------------------------
extern "C" void kernel_launch(void* const* d_in, const int* in_sizes, int n_in,
                              void* d_out, int out_size, void* d_ws, size_t ws_size,
                              hipStream_t stream) {
  (void)in_sizes; (void)n_in; (void)out_size; (void)ws_size;
  const float* obs = (const float*)d_in[0];
  const float* W1  = (const float*)d_in[1];  const float* b1  = (const float*)d_in[2];
  const float* W2  = (const float*)d_in[3];  const float* b2  = (const float*)d_in[4];
  const float* Wk1 = (const float*)d_in[5];  const float* bk1 = (const float*)d_in[6];
  const float* Wk2 = (const float*)d_in[7];  const float* bk2 = (const float*)d_in[8];
  const float* Wv1 = (const float*)d_in[9];  const float* bv1 = (const float*)d_in[10];
  const float* Wv2 = (const float*)d_in[11]; const float* bv2 = (const float*)d_in[12];
  const float* Wq1 = (const float*)d_in[13]; const float* bq1 = (const float*)d_in[14];
  const float* Wq2 = (const float*)d_in[15]; const float* bq2 = (const float*)d_in[16];
  const float* Wa1 = (const float*)d_in[17]; const float* ba1 = (const float*)d_in[18];
  const float* Wa2 = (const float*)d_in[19]; const float* ba2 = (const float*)d_in[20];

  char* ws = (char*)d_ws;
  size_t off = 0;
  auto alloc = [&](size_t elems) {
    bf16* p = (bf16*)(ws + off);
    off += (elems * sizeof(bf16) + 255) & ~(size_t)255;
    return p;
  };
  bf16* Wt1  = alloc(256 * 64);
  bf16* Wt2  = alloc(256 * 256);
  bf16* Wk1t = alloc(256 * 256);
  bf16* Wk2t = alloc(32 * 256);
  bf16* Wv1t = alloc(256 * 256);
  bf16* Wv2t = alloc(32 * 256);
  bf16* Wq1t = alloc(256 * 256);
  bf16* Wq2t = alloc(32 * 256);
  bf16* Wa1t = alloc(256 * 288);
  bf16* Wa2t = alloc(16 * 256);
  const size_t R = 512 * 256;
  bf16* xbf    = alloc(R * 256);
  bf16* kbf    = alloc(R * 32);
  bf16* qbf    = alloc(R * 32);
  bf16* vbf    = alloc(R * 32);
  bf16* commbf = alloc(R * 32);

  auto tr = [&](const float* W, bf16* Wt, int inD, int outD) {
    int total = inD * outD;
    k_transpose_bf16<<<(total + 255) / 256, 256, 0, stream>>>(W, Wt, inD, outD);
  };
  tr(W1, Wt1, 64, 256);    tr(W2, Wt2, 256, 256);
  tr(Wk1, Wk1t, 256, 256); tr(Wk2, Wk2t, 256, 32);
  tr(Wv1, Wv1t, 256, 256); tr(Wv2, Wv2t, 256, 32);
  tr(Wq1, Wq1t, 256, 256); tr(Wq2, Wq2t, 256, 32);
  tr(Wa1, Wa1t, 288, 256); tr(Wa2, Wa2t, 256, 16);

  k_obs2hidden<<<(int)(R / 64), 128, 0, stream>>>(obs, Wt1, b1, Wt2, b2, xbf);
  k_proj<<<(int)(R / 64), 128, 0, stream>>>(xbf, Wk1t, bk1, Wk2t, bk2, kbf);
  k_proj<<<(int)(R / 64), 128, 0, stream>>>(xbf, Wv1t, bv1, Wv2t, bv2, vbf);
  k_proj<<<(int)(R / 64), 128, 0, stream>>>(xbf, Wq1t, bq1, Wq2t, bq2, qbf);
  k_attn<<<512 * 4, 128, 0, stream>>>(qbf, kbf, vbf, commbf);
  k_head<<<(int)(R / 32), 64, 0, stream>>>(xbf, commbf, Wa1t, ba1, Wa2t, ba2,
                                           (float*)d_out);
}
